// ChainPoolNet_34291018891288
// MI455X (gfx1250) — matrix operand order repaired and checked by reading, not compiled, not run on hardware
//
#include <hip/hip_runtime.h>
#include <hip/hip_bf16.h>
#include <math.h>

typedef __attribute__((ext_vector_type(16))) _Float16 v16h;
typedef __attribute__((ext_vector_type(8)))  _Float16 v8h;
typedef __attribute__((ext_vector_type(4)))  _Float16 v4h;
typedef __attribute__((ext_vector_type(8)))  float    v8f;
typedef __attribute__((ext_vector_type(4)))  float    v4f;

#define NB  32
#define NN  1024
#define FIN 128
#define HH  256
#define CC  64
#define NE  (NB*NN*16)

enum {
  F_TRANS_A  = 1,
  F_TRANS_B  = 2,
  F_DIVDEG   = 4,
  F_ADD_D    = 8,
  F_BIAS     = 16,
  F_RELU     = 32,
  F_ROWSCALE = 64,
  F_LINK     = 128
};

// ---------------------------------------------------------------------------
// Batched 32x32-tile WMMA GEMM: C[b] = op(A[b]) @ op(B[b]) with fused epilogue.
// Compile-time FLAGS; 4 waves/block, each wave one 16x16 tile via
// v_wmma_f32_16x16x32_f16. f32->f16 conversion during LDS staging with
// float4 global loads; LDS double-buffered, one barrier per K-step.
// As is m-major, Bt is n-major so each lane's fragment is contiguous LDS
// (2x ds_load_b128 per operand per WMMA).
// ---------------------------------------------------------------------------
template<int FLAGS>
__global__ __launch_bounds__(128) void gemm32_wmma(
    const float* __restrict__ A, const float* __restrict__ Bm, float* __restrict__ C,
    int M, int N, int K, int lda, int ldb, int ldc,
    long sA, long sB, long sC,
    const float* __restrict__ bias,
    const float* __restrict__ D, long sD,
    const float* __restrict__ deg, long sDeg,
    const float* __restrict__ rowScale, long sRS,
    float* __restrict__ lossBuf)
{
  __shared__ _Float16 As[2][32][40];   // [buf][m][k]
  __shared__ _Float16 Bt[2][32][40];   // [buf][n][k]
  __shared__ float red[128];

  const int tid = threadIdx.x;
  const int b   = blockIdx.z;
  const int tm  = blockIdx.y * 32;
  const int tn  = blockIdx.x * 32;
  const float* Ab = A  + (long)b * sA;
  const float* Bb = Bm + (long)b * sB;

  const int w     = tid >> 5;
  const int lane  = tid & 31;
  const int wm    = (w >> 1) * 16;
  const int wn    = (w & 1) * 16;
  const int l15   = lane & 15;
  const int lhalf = lane >> 4;
  const int mrow  = wm + l15;
  const int ncol  = wn + l15;

  v8f acc = {};

  auto stage = [&](int k0, int buf) {
    if (!(FLAGS & F_TRANS_A)) {
      // contiguous along k: float4 load, packed v4h LDS store
#pragma unroll
      for (int i = 0; i < 2; ++i) {
        int chunk = tid + i * 128;
        int r = chunk >> 3, c4 = (chunk & 7) << 2;
        const v4f f = *(const v4f*)(Ab + (long)(tm + r) * lda + (k0 + c4));
        v4h h = { (_Float16)f.x, (_Float16)f.y, (_Float16)f.z, (_Float16)f.w };
        *(v4h*)&As[buf][r][c4] = h;
      }
    } else {
      // contiguous along m: float4 load, scatter into k column
#pragma unroll
      for (int i = 0; i < 2; ++i) {
        int chunk = tid + i * 128;
        int c = chunk >> 3, r4 = (chunk & 7) << 2;
        const v4f f = *(const v4f*)(Ab + (long)(k0 + c) * lda + (tm + r4));
        As[buf][r4 + 0][c] = (_Float16)f.x;
        As[buf][r4 + 1][c] = (_Float16)f.y;
        As[buf][r4 + 2][c] = (_Float16)f.z;
        As[buf][r4 + 3][c] = (_Float16)f.w;
      }
    }
    if (!(FLAGS & F_TRANS_B)) {
      // contiguous along n: float4 load, scatter into n-major rows
#pragma unroll
      for (int i = 0; i < 2; ++i) {
        int chunk = tid + i * 128;
        int r = chunk >> 3, c4 = (chunk & 7) << 2;   // r = k, c = n
        const v4f f = *(const v4f*)(Bb + (long)(k0 + r) * ldb + (tn + c4));
        Bt[buf][c4 + 0][r] = (_Float16)f.x;
        Bt[buf][c4 + 1][r] = (_Float16)f.y;
        Bt[buf][c4 + 2][r] = (_Float16)f.z;
        Bt[buf][c4 + 3][r] = (_Float16)f.w;
      }
    } else {
      // contiguous along k: float4 load, packed v4h store
#pragma unroll
      for (int i = 0; i < 2; ++i) {
        int chunk = tid + i * 128;
        int c = chunk >> 3, r4 = (chunk & 7) << 2;   // c = n, r = k
        const v4f f = *(const v4f*)(Bb + (long)(tn + c) * ldb + (k0 + r4));
        v4h h = { (_Float16)f.x, (_Float16)f.y, (_Float16)f.z, (_Float16)f.w };
        *(v4h*)&Bt[buf][c][r4] = h;
      }
    }
  };

  auto compute = [&](int buf) {
    // A fragment: lane's halves are As[m][lhalf*8 .. +7] ++ As[m][16+lhalf*8 .. +7]
    const v8h alo = *(const v8h*)&As[buf][mrow][lhalf * 8];
    const v8h ahi = *(const v8h*)&As[buf][mrow][16 + lhalf * 8];
    // B fragment: lane's halves are Bt[n][lhalf*8 .. +7] ++ Bt[n][16+lhalf*8 .. +7]
    const v8h blo = *(const v8h*)&Bt[buf][ncol][lhalf * 8];
    const v8h bhi = *(const v8h*)&Bt[buf][ncol][16 + lhalf * 8];
    v16h af = __builtin_shufflevector(alo, ahi, 0,1,2,3,4,5,6,7,8,9,10,11,12,13,14,15);
    v16h bf = __builtin_shufflevector(blo, bhi, 0,1,2,3,4,5,6,7,8,9,10,11,12,13,14,15);
    acc = __builtin_amdgcn_wmma_f32_16x16x32_f16(false, af, false, bf,
                                                 (short)0, acc, false, false);
  };

  const int nk = K >> 5;     // all K here are multiples of 64 -> nk even
  stage(0, 0);
  for (int i = 0; i < nk; i += 2) {
    __syncthreads();
    if (i + 1 < nk) stage((i + 1) << 5, 1);
    compute(0);
    __syncthreads();
    if (i + 2 < nk) stage((i + 2) << 5, 0);
    compute(1);
  }

  // C/D layout: lane l, VGPR r -> row (l>>4)*8 + r, col l&15 (within 16x16)
  float linkLocal = 0.0f;
  const int gn = tn + wn + l15;
#pragma unroll
  for (int r = 0; r < 8; ++r) {
    int gm = tm + wm + (lhalf << 3) + r;
    float v = acc[r];
    if (FLAGS & F_DIVDEG)   v /= deg[(long)b * sDeg + gm];
    if (FLAGS & F_ADD_D)    v += D[(long)b * sD + (long)gm * ldc + gn];
    if (FLAGS & F_BIAS)     v += bias[gn];
    if (FLAGS & F_RELU)     v = v > 0.f ? v : 0.f;
    if (FLAGS & F_ROWSCALE) v *= rowScale[(long)b * sRS + gm];
    if (FLAGS & F_LINK) {
      float dd = D[(long)b * sD + (long)gm * ldc + gn] - v;
      linkLocal += dd * dd;
    } else {
      C[(long)b * sC + (long)gm * ldc + gn] = v;
    }
  }

  if (FLAGS & F_LINK) {
    red[tid] = linkLocal;
    __syncthreads();
    for (int st = 64; st > 0; st >>= 1) {
      if (tid < st) red[tid] += red[tid + st];
      __syncthreads();
    }
    if (tid == 0) {
      long blk = ((long)blockIdx.z * gridDim.y + blockIdx.y) * gridDim.x + blockIdx.x;
      lossBuf[blk] = red[0];
    }
  }
}

// ---------------------------------------------------------------------------
// Helpers
// ---------------------------------------------------------------------------
__global__ void build_adj_kernel(const int* __restrict__ ei, float* __restrict__ adj)
{
  int e = blockIdx.x * blockDim.x + threadIdx.x;
  if (e >= NE) return;
  int e0 = ei[e];
  int e1 = ei[NE + e];
  int g   = e0 >> 10;       // / N
  int src = e0 & 1023;      // % N
  int dst = e1 & 1023;
  adj[((long)g << 20) + ((long)src << 10) + dst] = 1.0f;
}

__global__ void rowsum_clip_kernel(const float* __restrict__ adj,
                                   float* __restrict__ deg, int n)
{
  __shared__ float red[256];
  int tid = threadIdx.x;
  const float* row = adj + ((long)blockIdx.y * n + blockIdx.x) * n;
  float s = 0.f;
  for (int j = tid; j < n; j += 256) s += row[j];
  red[tid] = s; __syncthreads();
  for (int st = 128; st > 0; st >>= 1) {
    if (tid < st) red[tid] += red[tid + st];
    __syncthreads();
  }
  if (tid == 0) deg[(long)blockIdx.y * n + blockIdx.x] = fmaxf(red[0], 1.0f);
}

__global__ void l1norm64_kernel(const float* __restrict__ in, float* __restrict__ out)
{
  __shared__ float red[64];
  int row = blockIdx.x, tid = threadIdx.x;
  float v = in[(long)row * 64 + tid];
  red[tid] = fabsf(v); __syncthreads();
  for (int st = 32; st > 0; st >>= 1) {
    if (tid < st) red[tid] += red[tid + st];
    __syncthreads();
  }
  out[(long)row * 64 + tid] = v / fmaxf(red[0], 1e-12f);
}

// in-place row softmax + per-row entropy -> entBuf[row]
__global__ void softmax_ent_kernel(float* __restrict__ S, int w,
                                   float* __restrict__ entBuf)
{
  __shared__ float red[128];
  long row = blockIdx.x;
  float* p = S + row * (long)w;
  int tid = threadIdx.x;

  float mx = -3.4e38f;
  for (int j = tid; j < w; j += 128) mx = fmaxf(mx, p[j]);
  red[tid] = mx; __syncthreads();
  for (int st = 64; st > 0; st >>= 1) {
    if (tid < st) red[tid] = fmaxf(red[tid], red[tid + st]);
    __syncthreads();
  }
  mx = red[0]; __syncthreads();

  float sum = 0.f;
  for (int j = tid; j < w; j += 128) { float e = expf(p[j] - mx); p[j] = e; sum += e; }
  red[tid] = sum; __syncthreads();
  for (int st = 64; st > 0; st >>= 1) {
    if (tid < st) red[tid] += red[tid + st];
    __syncthreads();
  }
  sum = red[0]; __syncthreads();

  float inv = 1.f / sum, ent = 0.f;
  for (int j = tid; j < w; j += 128) {
    float s = p[j] * inv; p[j] = s;
    ent += -s * logf(s + 1e-15f);
  }
  red[tid] = ent; __syncthreads();
  for (int st = 64; st > 0; st >>= 1) {
    if (tid < st) red[tid] += red[tid + st];
    __syncthreads();
  }
  if (tid == 0) entBuf[row] = red[0];
}

// scale[b,c] = 1 / max(||c[:,c]||_2, eps) for c matrices [n,64] per batch
__global__ void colnorm_inv_kernel(const float* __restrict__ c,
                                   float* __restrict__ scale, int n)
{
  __shared__ float red[128];
  int col = blockIdx.x, b = blockIdx.y, tid = threadIdx.x;
  const float* base = c + (long)b * n * 64;
  float s = 0.f;
  for (int j = tid; j < n; j += 128) {
    float v = base[(long)j * 64 + col];
    s += v * v;
  }
  red[tid] = s; __syncthreads();
  for (int st = 64; st > 0; st >>= 1) {
    if (tid < st) red[tid] += red[tid + st];
    __syncthreads();
  }
  if (tid == 0) scale[(long)b * 64 + col] = 1.f / fmaxf(sqrtf(red[0]), 1e-12f);
}

// deterministic single-block reduction
__global__ void reduce_sum_kernel(const float* __restrict__ in, int n,
                                  float* __restrict__ out)
{
  __shared__ float red[256];
  int tid = threadIdx.x;
  float s = 0.f;
  for (int j = tid; j < n; j += 256) s += in[j];
  red[tid] = s; __syncthreads();
  for (int st = 128; st > 0; st >>= 1) {
    if (tid < st) red[tid] += red[tid + st];
    __syncthreads();
  }
  if (tid == 0) out[0] = red[0];
}

__global__ void finalize_loss_kernel(const float* __restrict__ acc,
                                     float* __restrict__ out)
{
  if (threadIdx.x == 0 && blockIdx.x == 0) {
    float loss = 0.f;
    loss += sqrtf(acc[0]) / (float)((long)NB * 1024 * 1024) + acc[3] / (float)(NB * 1024);
    loss += sqrtf(acc[1]) / (float)((long)NB * 512  * 512 ) + acc[4] / (float)(NB * 512 );
    loss += sqrtf(acc[2]) / (float)((long)NB * 256  * 256 ) + acc[5] / (float)(NB * 256 );
    out[4 * NB * CC * HH] = loss;
  }
}

// ---------------------------------------------------------------------------
static void gemm(hipStream_t st,
    const float* A, const float* B, float* C,
    int M, int N, int K, int lda, int ldb, int ldc,
    long sA, long sB, long sC, int flags,
    const float* bias = nullptr,
    const float* D = nullptr, long sD = 0,
    const float* deg = nullptr, long sDeg = 0,
    const float* rs = nullptr, long sRS = 0,
    float* lossBuf = nullptr)
{
  dim3 g(N / 32, M / 32, NB);
#define LG(F) gemm32_wmma<(F)><<<g, 128, 0, st>>>(A, B, C, M, N, K, lda, ldb, ldc, \
              sA, sB, sC, bias, D, sD, deg, sDeg, rs, sRS, lossBuf)
  switch (flags) {
    case 0:                            LG(0); break;
    case F_DIVDEG:                     LG(F_DIVDEG); break;
    case F_ADD_D | F_BIAS:             LG(F_ADD_D | F_BIAS); break;
    case F_ADD_D | F_BIAS | F_RELU:    LG(F_ADD_D | F_BIAS | F_RELU); break;
    case F_TRANS_A:                    LG(F_TRANS_A); break;
    case F_TRANS_A | F_ROWSCALE:       LG(F_TRANS_A | F_ROWSCALE); break;
    case F_TRANS_B | F_LINK:           LG(F_TRANS_B | F_LINK); break;
    default: break;
  }
#undef LG
}

extern "C" void kernel_launch(void* const* d_in, const int* in_sizes, int n_in,
                              void* d_out, int out_size, void* d_ws, size_t ws_size,
                              hipStream_t stream)
{
  const float* x   = (const float*)d_in[0];
  const float* cht = (const float*)d_in[1];
  const float* W1r = (const float*)d_in[2];
  const float* W1o = (const float*)d_in[3];
  const float* b1  = (const float*)d_in[4];
  const float* W2r = (const float*)d_in[5];
  const float* W2o = (const float*)d_in[6];
  const float* b2  = (const float*)d_in[7];
  const float* W3r = (const float*)d_in[8];
  const float* W3o = (const float*)d_in[9];
  const float* b3  = (const float*)d_in[10];
  const float* P1r = (const float*)d_in[11];
  const float* P1o = (const float*)d_in[12];
  const float* q1  = (const float*)d_in[13];
  const float* P2r = (const float*)d_in[14];
  const float* P2o = (const float*)d_in[15];
  const float* q2  = (const float*)d_in[16];
  const float* P3r = (const float*)d_in[17];
  const float* P3o = (const float*)d_in[18];
  const float* q3  = (const float*)d_in[19];
  const int*   ei  = (const int*)d_in[20];

  float* out = (float*)d_out;
  float* ws  = (float*)d_ws;
  size_t off = 0;
  auto alloc = [&](size_t n) { float* p = ws + off; off += n; return p; };

  float* adj0 = alloc((size_t)NB * NN * NN);     // [32,1024,1024]
  float* x1   = alloc((size_t)NB * NN * HH);     // [32,1024,256]
  float* cd0  = alloc((size_t)NB * NN * CC);     // [32,1024,64]
  float* s1   = alloc((size_t)NB * NN * 512);
  float* bufT = alloc((size_t)NB * NN * 512);
  float* bufA = alloc((size_t)NB * NN * HH);
  float* adj1 = alloc((size_t)NB * 512 * 512);
  float* xp1  = alloc((size_t)NB * 512 * HH);
  float* c1   = alloc((size_t)NB * 512 * CC);
  float* x2   = alloc((size_t)NB * 512 * HH);
  float* s2   = alloc((size_t)NB * 512 * 256);
  float* adj2 = alloc((size_t)NB * 256 * 256);
  float* xp2  = alloc((size_t)NB * 256 * HH);
  float* c2   = alloc((size_t)NB * 256 * CC);
  float* x3   = alloc((size_t)NB * 256 * HH);
  float* s3   = alloc((size_t)NB * 256 * 128);
  float* xp3  = alloc((size_t)NB * 128 * HH);
  float* c3   = alloc((size_t)NB * 128 * CC);
  float* deg0 = alloc((size_t)NB * NN);
  float* deg1 = alloc((size_t)NB * 512);
  float* deg2 = alloc((size_t)NB * 256);
  float* csc  = alloc((size_t)NB * CC);
  float* entB = alloc((size_t)NB * NN);
  float* lnkB = alloc((size_t)32768);
  float* acc  = alloc(8);
  (void)ws_size; (void)in_sizes; (void)n_in; (void)out_size;

  const long sEmb = (long)CC * HH;               // 16384 per-batch emb stride

  // ---- adjacency -----------------------------------------------------------
  hipMemsetAsync(adj0, 0, (size_t)NB * NN * NN * sizeof(float), stream);
  build_adj_kernel<<<NE / 256, 256, 0, stream>>>(ei, adj0);
  rowsum_clip_kernel<<<dim3(NN, NB), 256, 0, stream>>>(adj0, deg0, NN);
  l1norm64_kernel<<<NB * NN, 64, 0, stream>>>(cht, cd0);

  // ---- level 1 SAGE: x1 = relu((adj@x)/deg @ W1r + x@W1o + b1) -------------
  gemm(stream, adj0, x, bufA, NN, FIN, NN, NN, FIN, FIN,
       (long)NN * NN, (long)NN * FIN, (long)NN * FIN,
       F_DIVDEG, nullptr, nullptr, 0, deg0, NN);
  gemm(stream, bufA, W1r, x1, NN, HH, FIN, FIN, HH, HH,
       (long)NN * FIN, 0, (long)NN * HH, 0);
  gemm(stream, x, W1o, x1, NN, HH, FIN, FIN, HH, HH,
       (long)NN * FIN, 0, (long)NN * HH,
       F_ADD_D | F_BIAS | F_RELU, b1, x1, (long)NN * HH);

  // c_emb0 = cd0^T @ x1  -> out level 0
  gemm(stream, cd0, x1, out + 0 * (long)NB * sEmb, CC, HH, NN, CC, HH, HH,
       (long)NN * CC, (long)NN * HH, sEmb, F_TRANS_A);

  // ---- diff_pool level 1 (1024 -> 512) ------------------------------------
  gemm(stream, adj0, x1, bufA, NN, HH, NN, NN, HH, HH,
       (long)NN * NN, (long)NN * HH, (long)NN * HH,
       F_DIVDEG, nullptr, nullptr, 0, deg0, NN);
  gemm(stream, bufA, P1r, s1, NN, 512, HH, HH, 512, 512,
       (long)NN * HH, 0, (long)NN * 512, 0);
  gemm(stream, x1, P1o, s1, NN, 512, HH, HH, 512, 512,
       (long)NN * HH, 0, (long)NN * 512,
       F_ADD_D | F_BIAS, q1, s1, (long)NN * 512);
  softmax_ent_kernel<<<NB * NN, 128, 0, stream>>>(s1, 512, entB);
  reduce_sum_kernel<<<1, 256, 0, stream>>>(entB, NB * NN, acc + 3);

  gemm(stream, s1, x1, xp1, 512, HH, NN, 512, HH, HH,
       (long)NN * 512, (long)NN * HH, (long)512 * HH, F_TRANS_A);
  gemm(stream, adj0, s1, bufT, NN, 512, NN, NN, 512, 512,
       (long)NN * NN, (long)NN * 512, (long)NN * 512, 0);
  gemm(stream, s1, bufT, adj1, 512, 512, NN, 512, 512, 512,
       (long)NN * 512, (long)NN * 512, (long)512 * 512, F_TRANS_A);
  // link1: sum (adj0 - s1@s1^T)^2 -> lnkB (32*32*32 blocks)
  gemm(stream, s1, s1, bufT, NN, NN, 512, 512, 512, NN,
       (long)NN * 512, (long)NN * 512, 0,
       F_TRANS_B | F_LINK, nullptr, adj0, (long)NN * NN,
       nullptr, 0, nullptr, 0, lnkB);
  reduce_sum_kernel<<<1, 256, 0, stream>>>(lnkB, 32 * 32 * NB, acc + 0);

  gemm(stream, s1, cd0, c1, 512, CC, NN, 512, CC, CC,
       (long)NN * 512, (long)NN * CC, (long)512 * CC, F_TRANS_A);
  colnorm_inv_kernel<<<dim3(CC, NB), 128, 0, stream>>>(c1, csc, 512);
  gemm(stream, c1, xp1, out + 1 * (long)NB * sEmb, CC, HH, 512, CC, HH, HH,
       (long)512 * CC, (long)512 * HH, sEmb,
       F_TRANS_A | F_ROWSCALE, nullptr, nullptr, 0, nullptr, 0, csc, CC);

  // ---- level 2 SAGE + pool (512 -> 256) -----------------------------------
  rowsum_clip_kernel<<<dim3(512, NB), 256, 0, stream>>>(adj1, deg1, 512);
  gemm(stream, adj1, xp1, bufA, 512, HH, 512, 512, HH, HH,
       (long)512 * 512, (long)512 * HH, (long)512 * HH,
       F_DIVDEG, nullptr, nullptr, 0, deg1, 512);
  gemm(stream, bufA, W2r, x2, 512, HH, HH, HH, HH, HH,
       (long)512 * HH, 0, (long)512 * HH, 0);
  gemm(stream, xp1, W2o, x2, 512, HH, HH, HH, HH, HH,
       (long)512 * HH, 0, (long)512 * HH,
       F_ADD_D | F_BIAS | F_RELU, b2, x2, (long)512 * HH);

  gemm(stream, adj1, x2, bufA, 512, HH, 512, 512, HH, HH,
       (long)512 * 512, (long)512 * HH, (long)512 * HH,
       F_DIVDEG, nullptr, nullptr, 0, deg1, 512);
  gemm(stream, bufA, P2r, s2, 512, 256, HH, HH, 256, 256,
       (long)512 * HH, 0, (long)512 * 256, 0);
  gemm(stream, x2, P2o, s2, 512, 256, HH, HH, 256, 256,
       (long)512 * HH, 0, (long)512 * 256,
       F_ADD_D | F_BIAS, q2, s2, (long)512 * 256);
  softmax_ent_kernel<<<NB * 512, 128, 0, stream>>>(s2, 256, entB);
  reduce_sum_kernel<<<1, 256, 0, stream>>>(entB, NB * 512, acc + 4);

  gemm(stream, s2, x2, xp2, 256, HH, 512, 256, HH, HH,
       (long)512 * 256, (long)512 * HH, (long)256 * HH, F_TRANS_A);
  gemm(stream, adj1, s2, bufT, 512, 256, 512, 512, 256, 256,
       (long)512 * 512, (long)512 * 256, (long)512 * 256, 0);
  gemm(stream, s2, bufT, adj2, 256, 256, 512, 256, 256, 256,
       (long)512 * 256, (long)512 * 256, (long)256 * 256, F_TRANS_A);
  gemm(stream, s2, s2, bufT, 512, 512, 256, 256, 256, 512,
       (long)512 * 256, (long)512 * 256, 0,
       F_TRANS_B | F_LINK, nullptr, adj1, (long)512 * 512,
       nullptr, 0, nullptr, 0, lnkB);
  reduce_sum_kernel<<<1, 256, 0, stream>>>(lnkB, 16 * 16 * NB, acc + 1);

  gemm(stream, s2, c1, c2, 256, CC, 512, 256, CC, CC,
       (long)512 * 256, (long)512 * CC, (long)256 * CC, F_TRANS_A);
  colnorm_inv_kernel<<<dim3(CC, NB), 128, 0, stream>>>(c2, csc, 256);
  gemm(stream, c2, xp2, out + 2 * (long)NB * sEmb, CC, HH, 256, CC, HH, HH,
       (long)256 * CC, (long)256 * HH, sEmb,
       F_TRANS_A | F_ROWSCALE, nullptr, nullptr, 0, nullptr, 0, csc, CC);

  // ---- level 3 SAGE + pool (256 -> 128) -----------------------------------
  rowsum_clip_kernel<<<dim3(256, NB), 256, 0, stream>>>(adj2, deg2, 256);
  gemm(stream, adj2, xp2, bufA, 256, HH, 256, 256, HH, HH,
       (long)256 * 256, (long)256 * HH, (long)256 * HH,
       F_DIVDEG, nullptr, nullptr, 0, deg2, 256);
  gemm(stream, bufA, W3r, x3, 256, HH, HH, HH, HH, HH,
       (long)256 * HH, 0, (long)256 * HH, 0);
  gemm(stream, xp2, W3o, x3, 256, HH, HH, HH, HH, HH,
       (long)256 * HH, 0, (long)256 * HH,
       F_ADD_D | F_BIAS | F_RELU, b3, x3, (long)256 * HH);

  gemm(stream, adj2, x3, bufA, 256, HH, 256, 256, HH, HH,
       (long)256 * 256, (long)256 * HH, (long)256 * HH,
       F_DIVDEG, nullptr, nullptr, 0, deg2, 256);
  gemm(stream, bufA, P3r, s3, 256, 128, HH, HH, 128, 128,
       (long)256 * HH, 0, (long)256 * 128, 0);
  gemm(stream, x3, P3o, s3, 256, 128, HH, HH, 128, 128,
       (long)256 * HH, 0, (long)256 * 128,
       F_ADD_D | F_BIAS, q3, s3, (long)256 * 128);
  softmax_ent_kernel<<<NB * 256, 128, 0, stream>>>(s3, 128, entB);
  reduce_sum_kernel<<<1, 256, 0, stream>>>(entB, NB * 256, acc + 5);

  gemm(stream, s3, x3, xp3, 128, HH, 256, 128, HH, HH,
       (long)256 * 128, (long)256 * HH, (long)128 * HH, F_TRANS_A);
  gemm(stream, s3, s3, bufT, 256, 256, 128, 128, 128, 256,
       (long)256 * 128, (long)256 * 128, 0,
       F_TRANS_B | F_LINK, nullptr, adj2, (long)256 * 256,
       nullptr, 0, nullptr, 0, lnkB);
  reduce_sum_kernel<<<1, 256, 0, stream>>>(lnkB, 8 * 8 * NB, acc + 2);

  gemm(stream, s3, c2, c3, 128, CC, 256, 128, CC, CC,
       (long)256 * 128, (long)256 * CC, (long)128 * CC, F_TRANS_A);
  colnorm_inv_kernel<<<dim3(CC, NB), 128, 0, stream>>>(c3, csc, 128);
  gemm(stream, c3, xp3, out + 3 * (long)NB * sEmb, CC, HH, 128, CC, HH, HH,
       (long)128 * CC, (long)128 * HH, sEmb,
       F_TRANS_A | F_ROWSCALE, nullptr, nullptr, 0, nullptr, 0, csc, CC);

  // ---- total loss ----------------------------------------------------------
  finalize_loss_kernel<<<1, 32, 0, stream>>>(acc, out);
}
